// T5CrossAttention_21732534518218
// MI455X (gfx1250) — matrix operand order repaired and checked by reading, not compiled, not run on hardware
//
#include <hip/hip_runtime.h>
#include <stdint.h>

#define BATCH   2
#define QLEN    2048
#define KVLEN   2048
#define DMODEL  1024
#define NHEADS  16
#define DKV     64
#define EPSV    1e-6f
#define MROWS   (BATCH * QLEN)   // 4096

typedef unsigned short ushort_t;
typedef __attribute__((ext_vector_type(16))) __bf16 v16bf;
typedef __attribute__((ext_vector_type(8)))  __bf16 v8bf;
typedef __attribute__((ext_vector_type(8)))  float  v8f;

static __device__ __forceinline__ ushort_t f2bf(float f) {
    union { float f; uint32_t u; } cv; cv.f = f;
    uint32_t u = cv.u;
    uint32_t r = (u + 0x7FFFu + ((u >> 16) & 1u)) >> 16;
    return (ushort_t)r;
}

static __device__ __forceinline__ v16bf load16(const ushort_t* p) {
    return *reinterpret_cast<const v16bf*>(p);
}
static __device__ __forceinline__ v8bf load8(const ushort_t* p) {
    return *reinterpret_cast<const v8bf*>(p);
}
static __device__ __forceinline__ v16bf combine8(v8bf lo, v8bf hi) {
    v16bf r;
#pragma unroll
    for (int i = 0; i < 8; ++i) { r[i] = lo[i]; r[8 + i] = hi[i]; }
    return r;
}
static __device__ __forceinline__ v16bf loadA(const ushort_t* arow, int k0, int half) {
    return combine8(load8(arow + k0 + half * 16), load8(arow + k0 + half * 16 + 8));
}

// ---------------------------------------------------------------------------
// RMSNorm (T5 style) + cast to bf16.  One block per token row.
// ---------------------------------------------------------------------------
__global__ __launch_bounds__(256)
void rmsnorm_cast_kernel(const float* __restrict__ x,
                         const float* __restrict__ w,
                         ushort_t* __restrict__ out) {
    const int row = blockIdx.x;
    const float* xr = x + (size_t)row * DMODEL;
    __shared__ float red[8];

    float ss = 0.f;
    for (int i = threadIdx.x; i < DMODEL; i += blockDim.x) {
        float v = xr[i];
        ss += v * v;
    }
#pragma unroll
    for (int off = 16; off > 0; off >>= 1) ss += __shfl_down(ss, off, 32);
    const int lane = threadIdx.x & 31, wid = threadIdx.x >> 5;
    if (lane == 0) red[wid] = ss;
    __syncthreads();
    if (wid == 0) {
        float v = (lane < 8) ? red[lane] : 0.f;
#pragma unroll
        for (int off = 4; off > 0; off >>= 1) v += __shfl_down(v, off, 32);
        if (lane == 0) red[0] = v;
    }
    __syncthreads();
    const float inv = rsqrtf(red[0] / (float)DMODEL + EPSV);
    ushort_t* orow = out + (size_t)row * DMODEL;
    for (int i = threadIdx.x; i < DMODEL; i += blockDim.x)
        orow[i] = f2bf(xr[i] * inv * w[i]);
}

// ---------------------------------------------------------------------------
// Elementwise f32 -> bf16 cast.
// ---------------------------------------------------------------------------
__global__ __launch_bounds__(256)
void cast_bf16_kernel(const float* __restrict__ x, ushort_t* __restrict__ out, int n) {
    int i = blockIdx.x * blockDim.x + threadIdx.x;
    if (i < n) out[i] = f2bf(x[i]);
}

// ---------------------------------------------------------------------------
// Transpose-cast weight  w[K][N] (f32) -> wt[N][K] (bf16).  1024x1024.
// ---------------------------------------------------------------------------
__global__ __launch_bounds__(256)
void transpose_cast_kernel(const float* __restrict__ w, ushort_t* __restrict__ wt) {
    int idx = blockIdx.x * blockDim.x + threadIdx.x;
    int n = idx >> 10, k = idx & 1023;
    wt[(size_t)n * DMODEL + k] = f2bf(w[(size_t)k * DMODEL + n]);
}

// ---------------------------------------------------------------------------
// GEMM core: wave = 32x64 tile, 8 WMMA per 32-K step, branchless 2x-unrolled
// double-buffered fragment pipeline (no register copies, no inner branches).
// ---------------------------------------------------------------------------
static __device__ __forceinline__ void gemm_core_32x64(
        const ushort_t* __restrict__ arow0,
        const ushort_t* __restrict__ arow1,
        const ushort_t* __restrict__ W0,
        int half, v8f acc[2][4]) {
    v16bf a0[2], b0[4], a1[2], b1[4];

    a0[0] = loadA(arow0, 0, half);
    a0[1] = loadA(arow1, 0, half);
#pragma unroll
    for (int j = 0; j < 4; ++j) b0[j] = load16(W0 + (size_t)j * 16 * DMODEL);

    for (int k0 = 0; k0 < DMODEL; k0 += 64) {
        const int k1 = k0 + 32;                      // always <= DMODEL-32
        a1[0] = loadA(arow0, k1, half);
        a1[1] = loadA(arow1, k1, half);
#pragma unroll
        for (int j = 0; j < 4; ++j) b1[j] = load16(W0 + (size_t)j * 16 * DMODEL + k1);

#pragma unroll
        for (int rt = 0; rt < 2; ++rt)
#pragma unroll
            for (int j = 0; j < 4; ++j)
                acc[rt][j] = __builtin_amdgcn_wmma_f32_16x16x32_bf16(
                    false, a0[rt], false, b0[j], (short)0, acc[rt][j], false, false);

        const int k2 = (k0 + 64 < DMODEL) ? (k0 + 64) : 0;   // clamped (redundant last)
        a0[0] = loadA(arow0, k2, half);
        a0[1] = loadA(arow1, k2, half);
#pragma unroll
        for (int j = 0; j < 4; ++j) b0[j] = load16(W0 + (size_t)j * 16 * DMODEL + k2);

#pragma unroll
        for (int rt = 0; rt < 2; ++rt)
#pragma unroll
            for (int j = 0; j < 4; ++j)
                acc[rt][j] = __builtin_amdgcn_wmma_f32_16x16x32_bf16(
                    false, a1[rt], false, b1[j], (short)0, acc[rt][j], false, false);
    }
}

// ---------------------------------------------------------------------------
// Projection GEMM.  mode 0 -> store [bh][q][64] (Q,K); mode 1 -> [bh][64][kv] (V).
// ---------------------------------------------------------------------------
__global__ __launch_bounds__(128)
void gemm_qkv_kernel(const ushort_t* __restrict__ A,
                     const ushort_t* __restrict__ Wt,
                     ushort_t* __restrict__ dst,
                     int transpose_out) {
    const int wid  = threadIdx.x >> 5;
    const int lane = threadIdx.x & 31;
    const int nl   = lane & 15;
    const int half = lane >> 4;

    const int tile     = blockIdx.x * 4 + wid;     // (M/32)*(1024/64) = 2048 tiles
    const int row_tile = tile >> 4;
    const int col_tile = tile & 15;
    const int row0 = row_tile * 32;
    const int col0 = col_tile * 64;

    const ushort_t* arow0 = A + (size_t)(row0 + nl) * DMODEL;
    const ushort_t* arow1 = A + (size_t)(row0 + 16 + nl) * DMODEL;
    const ushort_t* W0    = Wt + (size_t)(col0 + nl) * DMODEL + half * 16;

    v8f acc[2][4] = {};
    gemm_core_32x64(arow0, arow1, W0, half, acc);

    const int b_idx = row0 / QLEN;     // 32-row tiles never straddle batch boundary
    const int h     = col_tile;
    const int bh    = b_idx * NHEADS + h;
    const int qbase = row0 % QLEN;
    if (transpose_out) {
#pragma unroll
        for (int rt = 0; rt < 2; ++rt)
#pragma unroll
            for (int j = 0; j < 4; ++j) {
                const int d = j * 16 + nl;
#pragma unroll
                for (int r = 0; r < 8; ++r) {
                    const int q = qbase + rt * 16 + half * 8 + r;
                    dst[((size_t)bh * DKV + d) * KVLEN + q] = f2bf(acc[rt][j][r]);
                }
            }
    } else {
#pragma unroll
        for (int rt = 0; rt < 2; ++rt)
#pragma unroll
            for (int j = 0; j < 4; ++j) {
                const int d = j * 16 + nl;
#pragma unroll
                for (int r = 0; r < 8; ++r) {
                    const int q = qbase + rt * 16 + half * 8 + r;
                    dst[((size_t)bh * QLEN + q) * DKV + d] = f2bf(acc[rt][j][r]);
                }
            }
    }
}

// ---------------------------------------------------------------------------
// Flash-style attention.  Block = 4 waves = 4 consecutive 16-row q tiles of
// ONE head.  K/V chunks double-buffered in LDS: next chunk is prefetched into
// registers at the top of each iteration and stored to the alternate buffer
// after compute -- one barrier per chunk, global latency hidden by compute.
// ---------------------------------------------------------------------------
__global__ __launch_bounds__(128)
void attention_kernel(const ushort_t* __restrict__ Qh,   // [B*H][QLEN][64]
                      const ushort_t* __restrict__ Kh,   // [B*H][KVLEN][64]
                      const ushort_t* __restrict__ Vt,   // [B*H][64][KVLEN]
                      ushort_t* __restrict__ ctx) {      // [B*QLEN][1024]
    __shared__ __align__(64) ushort_t Kch[2][32][64];    // K chunks [kv][d]   8KB
    __shared__ __align__(64) ushort_t Vch[2][64][32];    // V chunks [d][kv]   8KB
    __shared__ __align__(64) ushort_t Plds[4][16][32];   // per-wave P tile    4KB

    const int tid  = threadIdx.x;
    const int wid  = tid >> 5;
    const int lane = tid & 31;
    const int nl   = lane & 15;
    const int half = lane >> 4;

    const int bh = blockIdx.x / (QLEN / 64);             // one head per block
    const int q0 = (blockIdx.x % (QLEN / 64)) * 64 + wid * 16;

    const ushort_t* Qp = Qh + (size_t)bh * QLEN * DKV;
    const ushort_t* Kp = Kh + (size_t)bh * KVLEN * DKV;
    const ushort_t* Vp = Vt + (size_t)bh * DKV * KVLEN;

    // Q A-fragments for the two d k-steps (d 0..31, 32..63)
    v16bf qa[2];
#pragma unroll
    for (int ks = 0; ks < 2; ++ks)
        qa[ks] = loadA(Qp + (size_t)(q0 + nl) * DKV, ks * 32, half);

    float m_i[8], l_i[8];
#pragma unroll
    for (int r = 0; r < 8; ++r) { m_i[r] = -3.0e38f; l_i[r] = 0.f; }
    v8f acc[4] = {};

    const int kr = tid >> 2, kc = (tid & 3) * 16;        // K staging coords
    const int vr = tid >> 1, vc = (tid & 1) * 16;        // V staging coords

    // prologue: stage chunk 0 into buffer 0
    {
        v16bf kreg = load16(Kp + (size_t)kr * DKV + kc);
        v16bf vreg = load16(Vp + (size_t)vr * KVLEN + vc);
        *reinterpret_cast<v16bf*>(&Kch[0][kr][kc]) = kreg;
        *reinterpret_cast<v16bf*>(&Vch[0][vr][vc]) = vreg;
    }
    __syncthreads();

    int buf = 0;
    for (int kv0 = 0; kv0 < KVLEN; kv0 += 32) {
        // prefetch next chunk into registers (clamped; redundant on last iter)
        const int kvn = (kv0 + 32 < KVLEN) ? kv0 + 32 : 0;
        v16bf kreg = load16(Kp + (size_t)(kvn + kr) * DKV + kc);
        v16bf vreg = load16(Vp + (size_t)vr * KVLEN + kvn + vc);

        // scores: two 16(q) x 16(kv) tiles, K-dim = d = 64
        v8f s[2] = {};
#pragma unroll
        for (int nh = 0; nh < 2; ++nh)
#pragma unroll
            for (int ks = 0; ks < 2; ++ks) {
                v16bf kb = load16(&Kch[buf][nh * 16 + nl][ks * 32 + half * 16]);
                s[nh] = __builtin_amdgcn_wmma_f32_16x16x32_bf16(
                    false, qa[ks], false, kb, (short)0, s[nh], false, false);
            }

        // online softmax; lane owns rows m = half*8 + r
        float p0[8], p1[8];
#pragma unroll
        for (int r = 0; r < 8; ++r) {
            float vmax = fmaxf(s[0][r], s[1][r]);
#pragma unroll
            for (int off = 1; off < 16; off <<= 1)
                vmax = fmaxf(vmax, __shfl_xor(vmax, off, 32));
            const float m_new = fmaxf(m_i[r], vmax);
            const float scale = __expf(m_i[r] - m_new);
            p0[r] = __expf(s[0][r] - m_new);
            p1[r] = __expf(s[1][r] - m_new);
            float psum = p0[r] + p1[r];
#pragma unroll
            for (int off = 1; off < 16; off <<= 1)
                psum += __shfl_xor(psum, off, 32);
            l_i[r] = l_i[r] * scale + psum;
            m_i[r] = m_new;
#pragma unroll
            for (int j = 0; j < 4; ++j) acc[j][r] *= scale;
        }

        // stage P (C layout -> row-major 16x32 in this wave's LDS slice)
#pragma unroll
        for (int r = 0; r < 8; ++r) {
            Plds[wid][half * 8 + r][nl]      = f2bf(p0[r]);
            Plds[wid][half * 8 + r][16 + nl] = f2bf(p1[r]);
        }
        // intra-wave LDS RAW: ds ops from one wave execute in order
        const ushort_t* pp = &Plds[wid][nl][half * 16];
        v16bf pa = combine8(load8(pp), load8(pp + 8));
#pragma unroll
        for (int j = 0; j < 4; ++j) {
            v16bf vb = load16(&Vch[buf][j * 16 + nl][half * 16]);
            acc[j] = __builtin_amdgcn_wmma_f32_16x16x32_bf16(
                false, pa, false, vb, (short)0, acc[j], false, false);
        }

        // store prefetched chunk into the idle buffer; single barrier
        *reinterpret_cast<v16bf*>(&Kch[buf ^ 1][kr][kc]) = kreg;
        *reinterpret_cast<v16bf*>(&Vch[buf ^ 1][vr][vc]) = vreg;
        __syncthreads();
        buf ^= 1;
    }

    // normalize and store ctx (bf16, [b*Q + q][h*64 + d])
    const int b_idx = bh / NHEADS, h = bh % NHEADS;
    float inv_l[8];
#pragma unroll
    for (int r = 0; r < 8; ++r) inv_l[r] = 1.0f / l_i[r];
#pragma unroll
    for (int j = 0; j < 4; ++j) {
        const int d = j * 16 + nl;
#pragma unroll
        for (int r = 0; r < 8; ++r) {
            const int q = q0 + half * 8 + r;
            ctx[(size_t)(b_idx * QLEN + q) * DMODEL + h * DKV + d]
                = f2bf(acc[j][r] * inv_l[r]);
        }
    }
}

// ---------------------------------------------------------------------------
// Output projection + residual: out(f32) = ctx(bf16) @ wo + hidden.
// ---------------------------------------------------------------------------
__global__ __launch_bounds__(128)
void gemm_out_kernel(const ushort_t* __restrict__ A,    // ctx [4096][1024] bf16
                     const ushort_t* __restrict__ Wt,   // wo^T [1024][1024] bf16
                     const float* __restrict__ resid,   // hidden [4096][1024] f32
                     float* __restrict__ out) {
    const int wid  = threadIdx.x >> 5;
    const int lane = threadIdx.x & 31;
    const int nl   = lane & 15;
    const int half = lane >> 4;

    const int tile     = blockIdx.x * 4 + wid;
    const int row_tile = tile >> 4;
    const int col_tile = tile & 15;
    const int row0 = row_tile * 32;
    const int col0 = col_tile * 64;

    const ushort_t* arow0 = A + (size_t)(row0 + nl) * DMODEL;
    const ushort_t* arow1 = A + (size_t)(row0 + 16 + nl) * DMODEL;
    const ushort_t* W0    = Wt + (size_t)(col0 + nl) * DMODEL + half * 16;

    v8f acc[2][4] = {};
    gemm_core_32x64(arow0, arow1, W0, half, acc);

#pragma unroll
    for (int rt = 0; rt < 2; ++rt)
#pragma unroll
        for (int j = 0; j < 4; ++j)
#pragma unroll
            for (int r = 0; r < 8; ++r) {
                size_t idx = (size_t)(row0 + rt * 16 + half * 8 + r) * DMODEL
                           + col0 + j * 16 + nl;
                out[idx] = acc[rt][j][r] + resid[idx];
            }
}

// ---------------------------------------------------------------------------
extern "C" void kernel_launch(void* const* d_in, const int* in_sizes, int n_in,
                              void* d_out, int out_size, void* d_ws, size_t ws_size,
                              hipStream_t stream) {
    const float* hidden = (const float*)d_in[0];
    const float* kvs    = (const float*)d_in[1];
    const float* wq     = (const float*)d_in[2];
    const float* wk     = (const float*)d_in[3];
    const float* wv     = (const float*)d_in[4];
    const float* wo     = (const float*)d_in[5];
    const float* nw     = (const float*)d_in[6];
    float* out = (float*)d_out;

    char* ws = (char*)d_ws;
    size_t off = 0;
    auto alloc = [&](size_t bytes) -> void* {
        void* p = ws + off;
        off += (bytes + 255) & ~(size_t)255;
        return p;
    };

    const size_t act_bytes = (size_t)MROWS * DMODEL * 2;               // 8 MiB
    const size_t w_bytes   = (size_t)DMODEL * DMODEL * 2;              // 2 MiB
    const size_t hd_bytes  = (size_t)BATCH * NHEADS * QLEN * DKV * 2;  // 8 MiB

    ushort_t* normed = (ushort_t*)alloc(act_bytes);
    ushort_t* kvb    = (ushort_t*)alloc(act_bytes);
    ushort_t* wqt    = (ushort_t*)alloc(w_bytes);
    ushort_t* wkt    = (ushort_t*)alloc(w_bytes);
    ushort_t* wvt    = (ushort_t*)alloc(w_bytes);
    ushort_t* wot    = (ushort_t*)alloc(w_bytes);
    ushort_t* qbuf   = (ushort_t*)alloc(hd_bytes);
    ushort_t* kbuf   = (ushort_t*)alloc(hd_bytes);
    ushort_t* vbuf   = (ushort_t*)alloc(hd_bytes);   // transposed [bh][d][kv]
    ushort_t* ctxb   = (ushort_t*)alloc(act_bytes);

    // 1. RMSNorm + casts
    rmsnorm_cast_kernel<<<MROWS, 256, 0, stream>>>(hidden, nw, normed);
    {
        int n = BATCH * KVLEN * DMODEL;
        cast_bf16_kernel<<<(n + 255) / 256, 256, 0, stream>>>(kvs, kvb, n);
    }
    const int wgrid = (DMODEL * DMODEL) / 256;
    transpose_cast_kernel<<<wgrid, 256, 0, stream>>>(wq, wqt);
    transpose_cast_kernel<<<wgrid, 256, 0, stream>>>(wk, wkt);
    transpose_cast_kernel<<<wgrid, 256, 0, stream>>>(wv, wvt);
    transpose_cast_kernel<<<wgrid, 256, 0, stream>>>(wo, wot);

    // 2. Q/K/V projections (WMMA, 32x64 tiles, branchless pipeline)
    const int gemm_blocks = ((MROWS / 32) * (DMODEL / 64)) / 4;   // 512
    gemm_qkv_kernel<<<gemm_blocks, 128, 0, stream>>>(normed, wqt, qbuf, 0);
    gemm_qkv_kernel<<<gemm_blocks, 128, 0, stream>>>(kvb,    wkt, kbuf, 0);
    gemm_qkv_kernel<<<gemm_blocks, 128, 0, stream>>>(kvb,    wvt, vbuf, 1);

    // 3. Attention (WMMA flash-style, double-buffered LDS K/V chunks)
    const int attn_blocks = BATCH * NHEADS * (QLEN / 64);         // 1024
    attention_kernel<<<attn_blocks, 128, 0, stream>>>(qbuf, kbuf, vbuf, ctxb);

    // 4. Output projection + residual (WMMA, f32 epilogue)
    gemm_out_kernel<<<gemm_blocks, 128, 0, stream>>>(ctxb, wot, hidden, out);
}
